// multihead_self_attention_78288663871493
// MI455X (gfx1250) — matrix operand order repaired and checked by reading, compiled, not run on hardware
//
#include <hip/hip_runtime.h>
#include <math.h>

// Multi-head causal self-attention with RoPE for MI455X (gfx1250, wave32).
// All matmuls via v_wmma_f32_16x16x32_f16 (f16 operands, f32 accumulate).
#define B_SZ 2
#define S_LEN 2048
#define DM 1024
#define NH 16
#define DKH 64
#define MTOT (B_SZ * S_LEN) // 4096 tokens

typedef __attribute__((ext_vector_type(16))) _Float16 v16h;
typedef __attribute__((ext_vector_type(8)))  _Float16 v8h;
typedef __attribute__((ext_vector_type(8)))  float    v8f;
typedef __attribute__((ext_vector_type(4)))  int      v4i;

#if defined(__has_builtin)
#if __has_builtin(__builtin_amdgcn_global_load_async_to_lds_b128) && \
    __has_builtin(__builtin_amdgcn_s_wait_asynccnt)
#define HAVE_ASYNC_LDS 1
#endif
#endif
#ifndef HAVE_ASYNC_LDS
#define HAVE_ASYNC_LDS 0
#endif

#define GAS __attribute__((address_space(1)))
#define LAS __attribute__((address_space(3)))

__device__ __forceinline__ v8f wmma32(v16h a, v16h b, v8f c) {
  // D = A(16x32 f16) * B(32x16 f16) + C(16x16 f32)
  return __builtin_amdgcn_wmma_f32_16x16x32_f16(false, a, false, b, (short)0, c,
                                                false, false);
}

#if HAVE_ASYNC_LDS
// 16B async copy global -> LDS (tracked by ASYNCcnt).
__device__ __forceinline__ void async_cp16(const _Float16* gsrc, _Float16* ldst) {
  __builtin_amdgcn_global_load_async_to_lds_b128((GAS v4i*)gsrc, (LAS v4i*)ldst,
                                                 0, 0);
}
#endif

// A-fragment (16x32 f16, row-major source, leading dim ld):
// lane l: row m = l%16; halves 0..7 = K kb..kb+7, halves 8..15 = K kb+16..kb+23,
// kb = (l/16)*8  (per ISA 16-bit A layout). Two contiguous 16B loads.
__device__ __forceinline__ v16h load_a_frag(const _Float16* tile, int ld, int lane) {
  const int m  = lane & 15;
  const int kb = (lane >> 4) << 3;
  const _Float16* p = tile + (size_t)m * ld + kb;
  v16h r;
  ((v8h*)&r)[0] = *(const v8h*)(p);
  ((v8h*)&r)[1] = *(const v8h*)(p + 16);
  return r;
}

// B-fragment (32x16 f16) where B[k][n] = src[n*ld + k] (src holds B columns
// contiguously): lane l: col n = l%16; halves 0..7 = K kg..kg+7, halves
// 8..15 = K kg+16..kg+23, kg = (l/16)*8. Two contiguous 16B loads.
__device__ __forceinline__ v16h load_b_frag(const _Float16* tile, int ld, int lane) {
  const int n  = lane & 15;
  const int kg = (lane >> 4) << 3;
  const _Float16* p = tile + (size_t)n * ld + kg;
  v16h r;
  ((v8h*)&r)[0] = *(const v8h*)(p);
  ((v8h*)&r)[1] = *(const v8h*)(p + 16);
  return r;
}

__global__ void cvt_f32_f16(const float* __restrict__ src,
                            _Float16* __restrict__ dst, int n) {
  int i = blockIdx.x * blockDim.x + threadIdx.x;
  if (i < n) dst[i] = (_Float16)src[i];
}

// NT GEMM: C[m,n] = sum_k A[m,k] * W[n,k], M=MTOT, N=K=DM.
// Block = 4 waves computing a 64x64 tile; each wave owns 16 rows. The 64x32
// B k-panel is shared: staged into LDS with async-to-LDS (double-buffered)
// when available, else each wave loads B fragments straight from global.
// Modes:
//  0: Q  -> RoPE + 1/sqrt(dk), store f16 [b][h][s][d]
//  1: K  -> RoPE,              store f16 [b][h][s][d]
//  2: V  ->                    store f16 [b][h][d][s]  (transposed for PV wmma)
//  3: final projection ->      store f32 row-major [m][n] (d_out)
__global__ __launch_bounds__(128) void gemm_nt(const _Float16* __restrict__ A,
                                               const _Float16* __restrict__ W,
                                               void* __restrict__ out, int mode) {
  const int tid  = threadIdx.x;
  const int lane = tid & 31;
  const int wid  = tid >> 5;
  const int m0 = blockIdx.x * 64;
  const int n0 = blockIdx.y * 64;
  const int mw = m0 + wid * 16; // this wave's 16 rows
  v8f acc[4] = {};

#if HAVE_ASYNC_LDS
  // Double-buffered shared B panel: [buf][n (0..63)][k (0..31)] f16, 4KB/buf.
  // Cooperative stage: 4KB = 256 x 16B chunks; 128 threads x 2 chunks each.
  // Chunk c: n = c/4, k-halves = (c%4)*8  -> contiguous 16B of W row (n0+n).
  __shared__ __align__(16) _Float16 Bpan[2][64 * 32];
  {
    const int c0 = tid, c1 = tid + 128;
    const int na = c0 >> 2, ka = (c0 & 3) << 3;
    const int nb = c1 >> 2, kb = (c1 & 3) << 3;
    // prologue: stage k-slice 0 into buffer 0
    async_cp16(W + (size_t)(n0 + na) * DM + ka, &Bpan[0][na * 32 + ka]);
    async_cp16(W + (size_t)(n0 + nb) * DM + kb, &Bpan[0][nb * 32 + kb]);
    for (int s = 0; s < DM / 32; ++s) {
      const int buf = s & 1;
      if (s + 1 < DM / 32) {
        const int k1 = (s + 1) * 32;
        async_cp16(W + (size_t)(n0 + na) * DM + k1 + ka,
                   &Bpan[buf ^ 1][na * 32 + ka]);
        async_cp16(W + (size_t)(n0 + nb) * DM + k1 + kb,
                   &Bpan[buf ^ 1][nb * 32 + kb]);
        __builtin_amdgcn_s_wait_asynccnt(2); // stage s resident, s+1 in flight
      } else {
        __builtin_amdgcn_s_wait_asynccnt(0);
      }
      __syncthreads();
      v16h a = load_a_frag(A + (size_t)mw * DM + s * 32, DM, lane);
#pragma unroll
      for (int t = 0; t < 4; ++t) {
        v16h b = load_b_frag(&Bpan[buf][(t * 16) * 32], 32, lane);
        acc[t] = wmma32(a, b, acc[t]);
      }
      __syncthreads();
    }
  }
#else
  for (int k0 = 0; k0 < DM; k0 += 32) {
    v16h a = load_a_frag(A + (size_t)mw * DM + k0, DM, lane);
    if (k0 + 32 < DM)
      __builtin_prefetch(A + (size_t)(mw + (lane & 15)) * DM + k0 + 32, 0, 3);
    v16h b0 = load_b_frag(W + (size_t)(n0 +  0) * DM + k0, DM, lane);
    v16h b1 = load_b_frag(W + (size_t)(n0 + 16) * DM + k0, DM, lane);
    v16h b2 = load_b_frag(W + (size_t)(n0 + 32) * DM + k0, DM, lane);
    v16h b3 = load_b_frag(W + (size_t)(n0 + 48) * DM + k0, DM, lane);
    acc[0] = wmma32(a, b0, acc[0]);
    acc[1] = wmma32(a, b1, acc[1]);
    acc[2] = wmma32(a, b2, acc[2]);
    acc[3] = wmma32(a, b3, acc[3]);
  }
#endif

  // C layout: lane l, elem g -> row mw + g + 8*(l/16), col n0 + 16*t + (l%16)
  const int rowoff = (lane >> 4) << 3;
  const int colo   = lane & 15;
#pragma unroll
  for (int t = 0; t < 4; ++t) {
#pragma unroll
    for (int g = 0; g < 8; ++g) {
      int r = mw + g + rowoff;
      int c = n0 + t * 16 + colo;
      float v = acc[t][g];
      if (mode == 3) {
        ((float*)out)[(size_t)r * DM + c] = v;
      } else {
        int bb = r >> 11, s = r & (S_LEN - 1);
        int h = c >> 6, d = c & 63;
        _Float16* o16 = (_Float16*)out;
        if (mode == 2) {
          o16[((size_t)(bb * NH + h) * DKH + d) * S_LEN + s] = (_Float16)v;
        } else {
          // RoPE: pair (even,odd) columns live in adjacent lanes.
          float other = __shfl_xor(v, 1, 32);
          int j = d >> 1;
          float inv = __expf((float)j * (-2.0f / 64.0f) * 9.2103403719761836f);
          float ang = (float)s * inv;
          float cs = cosf(ang), sn = sinf(ang);
          float ro = (d & 1) ? (other * sn + v * cs) : (v * cs - other * sn);
          if (mode == 0) ro *= 0.125f; // 1/sqrt(64) folded into Q
          o16[((size_t)(bb * NH + h) * S_LEN + s) * DKH + d] = (_Float16)ro;
        }
      }
    }
  }
}

// Flash-attention: one wave per (batch*head, 16-query tile). Online softmax
// over 32-key blocks; P bounced through LDS to re-layout C-frag -> A-frag.
__global__ __launch_bounds__(32) void attn_kernel(const _Float16* __restrict__ Q,
                                                  const _Float16* __restrict__ K,
                                                  const _Float16* __restrict__ Vt,
                                                  _Float16* __restrict__ Ao) {
  __shared__ __align__(16) _Float16 Pbuf[16 * 32];
  const int lane = threadIdx.x;
  const int bh = blockIdx.x;
  const int qt = blockIdx.y;
  const int bb = bh / NH, h = bh % NH;
  const _Float16* Qb = Q  + (size_t)bh * S_LEN * DKH;
  const _Float16* Kb = K  + (size_t)bh * S_LEN * DKH;
  const _Float16* Vb = Vt + (size_t)bh * DKH * S_LEN; // [d][s]

  v16h q0 = load_a_frag(Qb + (size_t)qt * 16 * DKH + 0, DKH, lane);
  v16h q1 = load_a_frag(Qb + (size_t)qt * 16 * DKH + 32, DKH, lane);

  v8f o[4] = {};
  float mI[8], lI[8];
#pragma unroll
  for (int g = 0; g < 8; ++g) { mI[g] = -1e30f; lI[g] = 0.0f; }

  const int kmax = (qt + 1) * 16;          // causal horizon (exclusive)
  const int rowb = qt * 16 + ((lane >> 4) << 3);
  const int colb = lane & 15;

  for (int kb = 0; kb < kmax; kb += 32) {
    v8f s0 = {}, s1 = {};
    {
      v16h b0 = load_b_frag(Kb + (size_t)kb * DKH + 0, DKH, lane);
      v16h b1 = load_b_frag(Kb + (size_t)kb * DKH + 32, DKH, lane);
      s0 = wmma32(q0, b0, s0);
      s0 = wmma32(q1, b1, s0);
    }
    const bool st1 = (kb + 16) < kmax;     // wave-uniform
    if (st1) {
      v16h b0 = load_b_frag(Kb + (size_t)(kb + 16) * DKH + 0, DKH, lane);
      v16h b1 = load_b_frag(Kb + (size_t)(kb + 16) * DKH + 32, DKH, lane);
      s1 = wmma32(q0, b0, s1);
      s1 = wmma32(q1, b1, s1);
    }
#pragma unroll
    for (int g = 0; g < 8; ++g) {
      int r = rowb + g;
      float x0 = (kb + colb <= r) ? s0[g] : -1e30f;
      float x1 = (st1 && (kb + 16 + colb <= r)) ? s1[g] : -1e30f;
      float bm = fmaxf(x0, x1);
#pragma unroll
      for (int off = 1; off < 16; off <<= 1)   // row-max across 16-lane half
        bm = fmaxf(bm, __shfl_xor(bm, off, 32));
      float mNew = fmaxf(mI[g], bm);           // finite: diagonal always valid
      float sc = __expf(mI[g] - mNew);
      float p0 = __expf(x0 - mNew);
      float p1 = __expf(x1 - mNew);
      float ps = p0 + p1;
#pragma unroll
      for (int off = 1; off < 16; off <<= 1)   // row-sum
        ps += __shfl_xor(ps, off, 32);
      lI[g] = lI[g] * sc + ps;
      mI[g] = mNew;
#pragma unroll
      for (int t = 0; t < 4; ++t) o[t][g] *= sc;
      Pbuf[(g + ((lane >> 4) << 3)) * 32 + colb]      = (_Float16)p0;
      Pbuf[(g + ((lane >> 4) << 3)) * 32 + 16 + colb] = (_Float16)p1;
    }
    __syncthreads();
    v16h pa; // P as A-fragment (16x32), ld=32, from LDS
    {
      int m = lane & 15, kb2 = (lane >> 4) << 3;
      ((v8h*)&pa)[0] = *(const v8h*)&Pbuf[m * 32 + kb2];
      ((v8h*)&pa)[1] = *(const v8h*)&Pbuf[m * 32 + kb2 + 16];
    }
#pragma unroll
    for (int t = 0; t < 4; ++t) {
      v16h vb = load_b_frag(Vb + (size_t)(t * 16) * S_LEN + kb, S_LEN, lane);
      o[t] = wmma32(pa, vb, o[t]);
    }
    __syncthreads();
  }
  // Write attention output f16 row-major [b][s][h*64+d] for final projection.
#pragma unroll
  for (int g = 0; g < 8; ++g) {
    float invl = 1.0f / lI[g];
    int srow = rowb + g;
#pragma unroll
    for (int t = 0; t < 4; ++t) {
      int d = t * 16 + colb;
      Ao[((size_t)(bb * S_LEN + srow)) * DM + h * DKH + d] =
          (_Float16)(o[t][g] * invl);
    }
  }
}

extern "C" void kernel_launch(void* const* d_in, const int* in_sizes, int n_in,
                              void* d_out, int out_size, void* d_ws, size_t ws_size,
                              hipStream_t stream) {
  (void)in_sizes; (void)n_in; (void)out_size; (void)ws_size;
  const float* x  = (const float*)d_in[0];
  const float* Wq = (const float*)d_in[1];
  const float* Wk = (const float*)d_in[2];
  const float* Wv = (const float*)d_in[3];
  const float* Wo = (const float*)d_in[4];

  const size_t NX = (size_t)MTOT * DM; // 4M activations
  const size_t NW = (size_t)DM * DM;   // 1M per weight
  _Float16* ws  = (_Float16*)d_ws;     // ~48 MB total
  _Float16* xh  = ws;
  _Float16* wqh = xh  + NX;
  _Float16* wkh = wqh + NW;
  _Float16* wvh = wkh + NW;
  _Float16* woh = wvh + NW;
  _Float16* Qh  = woh + NW;            // [b][h][s][d]
  _Float16* Kh  = Qh  + NX;            // [b][h][s][d]
  _Float16* Vt  = Kh  + NX;            // [b][h][d][s]
  _Float16* Aoh = Vt  + NX;            // [b][s][D]

  cvt_f32_f16<<<dim3((unsigned)((NX + 255) / 256)), 256, 0, stream>>>(x, xh, (int)NX);
  cvt_f32_f16<<<dim3((unsigned)((NW + 255) / 256)), 256, 0, stream>>>(Wq, wqh, (int)NW);
  cvt_f32_f16<<<dim3((unsigned)((NW + 255) / 256)), 256, 0, stream>>>(Wk, wkh, (int)NW);
  cvt_f32_f16<<<dim3((unsigned)((NW + 255) / 256)), 256, 0, stream>>>(Wv, wvh, (int)NW);
  cvt_f32_f16<<<dim3((unsigned)((NW + 255) / 256)), 256, 0, stream>>>(Wo, woh, (int)NW);

  dim3 ggrid(MTOT / 64, DM / 64);
  gemm_nt<<<ggrid, 128, 0, stream>>>(xh, wqh, Qh, 0);
  gemm_nt<<<ggrid, 128, 0, stream>>>(xh, wkh, Kh, 1);
  gemm_nt<<<ggrid, 128, 0, stream>>>(xh, wvh, Vt, 2);

  attn_kernel<<<dim3(B_SZ * NH, S_LEN / 16), 32, 0, stream>>>(Qh, Kh, Vt, Aoh);

  gemm_nt<<<ggrid, 128, 0, stream>>>(Aoh, woh, d_out, 3);
}
